// FPDeformableEncTransformer_44702019617529
// MI455X (gfx1250) — matrix operand order repaired and checked by reading, not compiled, not run on hardware
//
#include <hip/hip_runtime.h>

#define DEVFN __device__ __forceinline__

typedef __attribute__((ext_vector_type(16))) __bf16   v16bf;
typedef __attribute__((ext_vector_type(8)))  float    v8f;
typedef __attribute__((ext_vector_type(8)))  unsigned v8u;
typedef __attribute__((ext_vector_type(4)))  unsigned v4u;
// TDM descriptor group types (6-arg builtin form on this toolchain).
typedef __attribute__((ext_vector_type(4)))  unsigned tdm_g0_t;
typedef __attribute__((ext_vector_type(8)))  int      tdm_g1_t;
typedef __attribute__((ext_vector_type(4)))  int      tdm_g2_t;

static constexpr int C_DIM  = 256;
static constexpr int N_B    = 4;
static constexpr int LQ     = 5440;
static constexpr int MTOK   = N_B * LQ;   // 21760 tokens (both query and value sides)
static constexpr int LEN_IN = 5440;

// ---------- bf16 helpers (RNE) ----------
DEVFN unsigned short f32_to_bf16(float f) {
  unsigned u = __float_as_uint(f);
  u += 0x7fffu + ((u >> 16) & 1u);
  return (unsigned short)(u >> 16);
}
DEVFN float bf16_to_f32(unsigned short h) {
  return __uint_as_float(((unsigned)h) << 16);
}

// Split fp32 plane into bf16 hi + bf16 residual (for bf16x3 GEMM).
__global__ void k_split(const float* __restrict__ x,
                        unsigned short* __restrict__ hi,
                        unsigned short* __restrict__ lo, int n) {
  int i = blockIdx.x * blockDim.x + threadIdx.x;
  if (i >= n) return;
  float v = x[i];
  unsigned short h = f32_to_bf16(v);
  hi[i] = h;
  lo[i] = f32_to_bf16(v - bf16_to_f32(h));
}

// Repack a K x Nc fp32 weight into WMMA 32x16 B-fragment order (hi & lo planes).
// dword id = ((kt*NT + tn)*32 + lane)*8 + v ; lane%16 = column, lane/16 selects K 0-7,16-23 / 8-15,24-31,
// dword = {bf16(w[k][n]), bf16(w[k+1][n])} with k = kt*32 + (lane/16)*8 + (v/4)*16 + 2(v%4).
__global__ void k_repack(const float* __restrict__ w, unsigned* __restrict__ pHi,
                         unsigned* __restrict__ pLo, int K, int Nc) {
  int id = blockIdx.x * blockDim.x + threadIdx.x;
  int total = (K * Nc) >> 1;
  if (id >= total) return;
  int v    = id & 7;
  int lane = (id >> 3) & 31;
  int tile = id >> 8;
  int NT = Nc >> 4;
  int tn = tile % NT;
  int kt = tile / NT;
  int nn = tn * 16 + (lane & 15);
  int k  = kt * 32 + ((lane >> 4) << 3) + ((v & 4) << 2) + ((v & 3) << 1);
  float w0 = w[(size_t)k * Nc + nn];
  float w1 = w[(size_t)(k + 1) * Nc + nn];
  unsigned short h0 = f32_to_bf16(w0), h1 = f32_to_bf16(w1);
  unsigned short l0 = f32_to_bf16(w0 - bf16_to_f32(h0));
  unsigned short l1 = f32_to_bf16(w1 - bf16_to_f32(h1));
  pHi[id] = (unsigned)h0 | ((unsigned)h1 << 16);
  pLo[id] = (unsigned)l0 | ((unsigned)l1 << 16);
}

// A fragment per ISA 16x32 bf16 layout: lane holds K {kh8..kh8+7} and {kh8+16..kh8+23}
// as two contiguous 16-byte chunks (works for both global and LDS pointers).
DEVFN v16bf ld_frag(const unsigned short* p) {
  union { v8u v8; v4u v4[2]; } u;
  u.v4[0] = *(const v4u*)p;
  u.v4[1] = *(const v4u*)(p + 16);
  return __builtin_bit_cast(v16bf, u.v8);
}

// C = A(f32 as bf16 hi/lo, MxK row-major) @ B(prepacked bf16 hi/lo) + bias.
// 8 waves/block = one 16-row strip x 128 output cols. The A strip (hi+lo) is staged
// into LDS by the Tensor Data Mover (2D 16xK bf16 tile, HW row padding of 16B to
// spread LDS banks), wave0 drains TENSORcnt, then all 8 waves feed WMMA fragments
// from ds_load_b128. bf16x3: hi*hi + hi*lo + lo*hi, f32 accum.
template <int K, int NT>
__global__ void __launch_bounds__(256)
k_gemm_bf16x3(const unsigned short* __restrict__ Ahi,
              const unsigned short* __restrict__ Alo,
              const unsigned* __restrict__ Bhi,
              const unsigned* __restrict__ Blo,
              const float* __restrict__ bias,
              float* __restrict__ Cout) {
  constexpr int Nc  = NT * 16;
  constexpr int LDK = K + 8;                 // padded LDS row stride (bf16 units)
  static_assert(K == 256, "pad_interval encoding below assumes 512B rows");
  __shared__ unsigned short lds[2][16 * LDK];

  int tid   = threadIdx.x;
  int lane  = tid & 31;
  int wv    = tid >> 5;
  int tileM = blockIdx.x;
  int tileN = blockIdx.y * 8 + wv;

#if __has_builtin(__builtin_amdgcn_tensor_load_to_lds)
  // ---- TDM stage of the 16 x K A strip (hi & lo planes) into LDS ----
  if (wv == 0) {
    // lds sits at LDS offset 0 (sole shared object); plane offsets relative to it.
    unsigned ldsH = 0u;
    unsigned ldsL = (unsigned)((const char*)&lds[1][0] - (const char*)&lds[0][0]);
    unsigned long long gaH = (unsigned long long)(Ahi + (size_t)tileM * 16 * K);
    unsigned long long gaL = (unsigned long long)(Alo + (size_t)tileM * 16 * K);
    // D# group0: [1:0]=count=1, [63:32]=lds_addr, [120:64]=global_addr, [127:126]=type 2
    tdm_g0_t g0h = { 1u, ldsH, (unsigned)gaH,
                     (unsigned)((gaH >> 32) & 0x1FFFFFFu) | (2u << 30) };
    tdm_g0_t g0l = { 1u, ldsL, (unsigned)gaL,
                     (unsigned)((gaL >> 32) & 0x1FFFFFFu) | (2u << 30) };
    // D# group1: data_size=1(2B), pad_enable=1, pad_interval=6 (128 DW = one 512B row),
    // pad_amount=3 (4 DW = 16B = 8 bf16 -> LDK = K+8);
    // word1[31:16] = tensor_dim0 lo16 = K, word2[15:0] = tensor_dim0 hi16 = 0,
    // word2[31:16] = tensor_dim1 lo16 = 16, word3[15:0] = tensor_dim1 hi16 = 0,
    // word3[31:16] = tile_dim0 = K, word4[15:0] = tile_dim1 = 16,
    // word5 = tensor_dim0_stride[31:0] = K, word6 = word7 = 0.
    v8u g1u = { (1u << 16) | (1u << 20) | (6u << 22) | (3u << 25),
                (unsigned)K << 16,
                16u << 16,
                (unsigned)K << 16,
                16u,
                (unsigned)K,
                0u, 0u };
    tdm_g1_t g1 = __builtin_bit_cast(tdm_g1_t, g1u);
    tdm_g2_t z4 = { 0, 0, 0, 0 };      // groups 2/3 unused (2-D tensor)
    tdm_g1_t z8 = { 0, 0, 0, 0, 0, 0, 0, 0 };
    __builtin_amdgcn_tensor_load_to_lds(g0h, g1, z4, z4, z8, 0);
    __builtin_amdgcn_tensor_load_to_lds(g0l, g1, z4, z4, z8, 0);
    __builtin_amdgcn_s_wait_tensorcnt((short)0);
  }
#else
  // ---- fallback: cooperative VGPR-mediated stage ----
  {
    int r  = tid >> 4;
    int kc = (tid & 15) * (K / 16);
    const unsigned short* gH = Ahi + ((size_t)tileM * 16 + r) * K + kc;
    const unsigned short* gL = Alo + ((size_t)tileM * 16 + r) * K + kc;
    unsigned short* sH = &lds[0][r * LDK + kc];
    unsigned short* sL = &lds[1][r * LDK + kc];
#pragma unroll
    for (int j = 0; j < K / 128; ++j) {
      *(v4u*)(sH + j * 8) = *(const v4u*)(gH + j * 8);
      *(v4u*)(sL + j * 8) = *(const v4u*)(gL + j * 8);
    }
  }
#endif
  __syncthreads();

  int row = lane & 15;
  int kh8 = (lane >> 4) << 3;
  const unsigned short* aH = &lds[0][row * LDK + kh8];
  const unsigned short* aL = &lds[1][row * LDK + kh8];
  const v8u* bHp = (const v8u*)Bhi + (size_t)tileN * 32 + lane;
  const v8u* bLp = (const v8u*)Blo + (size_t)tileN * 32 + lane;

  v8f c = {};
#pragma unroll
  for (int kt = 0; kt < K / 32; ++kt) {
    v16bf fH = ld_frag(aH + kt * 32);
    v16bf fL = ld_frag(aL + kt * 32);
    v16bf bH = __builtin_bit_cast(v16bf, bHp[(size_t)kt * NT * 32]);
    v16bf bL = __builtin_bit_cast(v16bf, bLp[(size_t)kt * NT * 32]);
    c = __builtin_amdgcn_wmma_f32_16x16x32_bf16(false, fL, false, bH, (short)0, c, false, false);
    c = __builtin_amdgcn_wmma_f32_16x16x32_bf16(false, fH, false, bL, (short)0, c, false, false);
    c = __builtin_amdgcn_wmma_f32_16x16x32_bf16(false, fH, false, bH, (short)0, c, false, false);
  }

  union { v8f v; float f[8]; } cu; cu.v = c;
  int nCol  = (tileN << 4) + (lane & 15);
  int mBase = (tileM << 4) + ((lane >> 4) << 3);
  float b = bias[nCol];
#pragma unroll
  for (int v = 0; v < 8; ++v)
    Cout[(size_t)(mBase + v) * Nc + nCol] = cu.f[v] + b;
}

// Softmax over the 16 (level,point) logits per (token,head). Rows are contiguous.
__global__ void k_softmax16(float* __restrict__ a, int rows) {
  int id = blockIdx.x * blockDim.x + threadIdx.x;
  if (id >= rows) return;
  float* p = a + (size_t)id * 16;
  float m = p[0];
#pragma unroll
  for (int i = 1; i < 16; ++i) m = fmaxf(m, p[i]);
  float e[16];
  float s = 0.f;
#pragma unroll
  for (int i = 0; i < 16; ++i) { e[i] = __expf(p[i] - m); s += e[i]; }
  float inv = 1.0f / s;
#pragma unroll
  for (int i = 0; i < 16; ++i) p[i] = e[i] * inv;
}

// Deformable sampling core: one wave per (token, head); lane = channel d (D=32 == wave32)
// so every corner fetch is one coalesced 128B transaction. Coordinate math is wave-uniform.
__global__ void __launch_bounds__(256)
k_sample(const float* __restrict__ value, const float* __restrict__ offs,
         const float* __restrict__ attn, const float* __restrict__ refp,
         float* __restrict__ core) {
  const int LW[4] = {64, 32, 16, 8};
  const int LS[4] = {0, 4096, 5120, 5376};
  int gw = blockIdx.x * 8 + (threadIdx.x >> 5);
  int d  = threadIdx.x & 31;
  int token = gw >> 3;
  int hd    = gw & 7;
  int n = token / LQ;
  const float* offRow = offs + (size_t)token * 256;  // (h,l,p,2)
  const float* attRow = attn + (size_t)token * 128;  // (h,l,p)
  const float* refRow = refp + (size_t)token * 8;    // (l,2)
  const float* vbase  = value + (size_t)n * LEN_IN * 256 + hd * 32 + d;
  float acc = 0.f;
#pragma unroll
  for (int lvl = 0; lvl < 4; ++lvl) {
    int W = LW[lvl], H = LW[lvl], s = LS[lvl];
    // pix = (ref + off/size)*size - 0.5 == ref*size + off - 0.5
    float rx = refRow[lvl * 2 + 0] * (float)W - 0.5f;
    float ry = refRow[lvl * 2 + 1] * (float)H - 0.5f;
#pragma unroll
    for (int p = 0; p < 4; ++p) {
      int base = (hd * 4 + lvl) * 4 + p;
      float x = rx + offRow[base * 2 + 0];
      float y = ry + offRow[base * 2 + 1];
      float a = attRow[base];
      float xf = floorf(x), yf = floorf(y);
      float fx = x - xf, fy = y - yf;
      int x0 = (int)xf, y0 = (int)yf;
      bool xi0 = (x0 >= 0) && (x0 < W);
      bool xi1 = (x0 + 1 >= 0) && (x0 + 1 < W);
      bool yi0 = (y0 >= 0) && (y0 < H);
      bool yi1 = (y0 + 1 >= 0) && (y0 + 1 < H);
      float g00 = 0.f, g10 = 0.f, g01 = 0.f, g11 = 0.f;
      if (yi0) {
        const float* row = vbase + (size_t)(s + y0 * W) * 256;
        if (xi0) g00 = row[(size_t)x0 * 256];
        if (xi1) g10 = row[(size_t)(x0 + 1) * 256];
      }
      if (yi1) {
        const float* row = vbase + (size_t)(s + (y0 + 1) * W) * 256;
        if (xi0) g01 = row[(size_t)x0 * 256];
        if (xi1) g11 = row[(size_t)(x0 + 1) * 256];
      }
      acc += a * ((g00 * (1.f - fx) + g10 * fx) * (1.f - fy) +
                  (g01 * (1.f - fx) + g11 * fx) * fy);
    }
  }
  core[(size_t)token * 256 + hd * 32 + d] = acc;
}

extern "C" void kernel_launch(void* const* d_in, const int* in_sizes, int n_in,
                              void* d_out, int out_size, void* d_ws, size_t ws_size,
                              hipStream_t stream) {
  (void)in_sizes; (void)n_in; (void)out_size; (void)ws_size;
  const float* query  = (const float*)d_in[0];
  const float* refp   = (const float*)d_in[1];
  const float* inflat = (const float*)d_in[2];
  // d_in[3]/d_in[4] (shapes, level starts) are compile-time constants here.
  const float* w_val  = (const float*)d_in[5];
  const float* b_val  = (const float*)d_in[6];
  const float* w_off  = (const float*)d_in[7];
  const float* b_off  = (const float*)d_in[8];
  const float* w_attn = (const float*)d_in[9];
  const float* b_attn = (const float*)d_in[10];
  const float* w_out  = (const float*)d_in[11];
  const float* b_out  = (const float*)d_in[12];

  const size_t SZ_BF = (size_t)MTOK * C_DIM * 2;  // one bf16 activation plane (11.14 MB)
  char* w = (char*)d_ws;
  unsigned short* inHi = (unsigned short*)(w);
  unsigned short* inLo = (unsigned short*)(w + SZ_BF);
  unsigned short* qHi  = (unsigned short*)(w + 2 * SZ_BF);
  unsigned short* qLo  = (unsigned short*)(w + 3 * SZ_BF);
  float* valueF = (float*)(w + 4 * SZ_BF);   // 2*SZ_BF bytes
  float* offsF  = (float*)(w + 6 * SZ_BF);   // 2*SZ_BF bytes
  float* attnF  = (float*)(w + 8 * SZ_BF);   // SZ_BF bytes
  char* wp = w + 9 * SZ_BF;                  // packed weights (~0.9 MB)
  unsigned* wvalHi  = (unsigned*)wp;
  unsigned* wvalLo  = wvalHi + 32768;
  unsigned* woffHi  = wvalLo + 32768;
  unsigned* woffLo  = woffHi + 32768;
  unsigned* wattnHi = woffLo + 32768;
  unsigned* wattnLo = wattnHi + 16384;
  unsigned* woutHi  = wattnLo + 16384;
  unsigned* woutLo  = woutHi + 32768;
  // Aliases: core overlays consumed input planes (safe: kernels serialize on stream).
  float* coreF = (float*)w;                  // overlays inHi+inLo (exactly 2*SZ_BF bytes)
  unsigned short* coreHi = qHi;              // query no longer needed after GEMM2/3
  unsigned short* coreLo = qLo;

  const int nAct = MTOK * C_DIM;             // 5,570,560 (divisible by 256)
  dim3 blk(256);

  k_split<<<dim3(nAct / 256), blk, 0, stream>>>(inflat, inHi, inLo, nAct);
  k_split<<<dim3(nAct / 256), blk, 0, stream>>>(query, qHi, qLo, nAct);
  k_repack<<<dim3(128), blk, 0, stream>>>(w_val,  wvalHi,  wvalLo,  256, 256);
  k_repack<<<dim3(128), blk, 0, stream>>>(w_off,  woffHi,  woffLo,  256, 256);
  k_repack<<<dim3(64),  blk, 0, stream>>>(w_attn, wattnHi, wattnLo, 256, 128);
  k_repack<<<dim3(128), blk, 0, stream>>>(w_out,  woutHi,  woutLo,  256, 256);

  dim3 g256(MTOK / 16, 2), g128(MTOK / 16, 1);
  k_gemm_bf16x3<256, 16><<<g256, blk, 0, stream>>>(inHi, inLo, wvalHi, wvalLo, b_val, valueF);
  k_gemm_bf16x3<256, 16><<<g256, blk, 0, stream>>>(qHi, qLo, woffHi, woffLo, b_off, offsF);
  k_gemm_bf16x3<256, 8><<<g128, blk, 0, stream>>>(qHi, qLo, wattnHi, wattnLo, b_attn, attnF);

  k_softmax16<<<dim3((MTOK * 8 + 255) / 256), blk, 0, stream>>>(attnF, MTOK * 8);
  k_sample<<<dim3(MTOK), blk, 0, stream>>>(valueF, offsF, attnF, refp, coreF);

  k_split<<<dim3(nAct / 256), blk, 0, stream>>>(coreF, coreHi, coreLo, nAct);
  k_gemm_bf16x3<256, 16><<<g256, blk, 0, stream>>>(coreHi, coreLo, woutHi, woutLo, b_out,
                                                   (float*)d_out);
}
